// GNNEnhancedNet_81252191306418
// MI455X (gfx1250) — compile-verified, matrix operand order checked by reading, not run on hardware
//
#include <hip/hip_runtime.h>
#include <hip/hip_bf16.h>

#define TT 4096
#define PPROV 2
#define NFEAT 16
#define HID 64
#define GOUT 32
#define FDIM (TT*PPROV)

typedef __attribute__((ext_vector_type(16))) __bf16 v16bf;
typedef __attribute__((ext_vector_type(8)))  float  v8f;
typedef int v4i __attribute__((vector_size(4 * sizeof(int))));

#define GAS1 __attribute__((address_space(1)))
#define LAS3 __attribute__((address_space(3)))

#if __has_builtin(__builtin_amdgcn_global_load_async_to_lds_b128)
#define HAVE_ASYNC_LDS 1
#endif

__device__ __forceinline__ void wait_async0() {
#if defined(HAVE_ASYNC_LDS)
#if __has_builtin(__builtin_amdgcn_s_wait_asynccnt)
    __builtin_amdgcn_s_wait_asynccnt(0);
#else
    asm volatile("s_wait_asynccnt 0x0" ::: "memory");
#endif
#endif
}

union Frag16 {
    v16bf v;
    unsigned short h[16];
    unsigned int u[8];
};

__device__ __forceinline__ unsigned short f2bf(float f) {
    unsigned x = __float_as_uint(f);
    unsigned r = x + 0x7FFFu + ((x >> 16) & 1u);  // round-to-nearest-even
    return (unsigned short)(r >> 16);
}

// ---------------------------------------------------------------------------
// WMMA GEMM: C[M,N] = act(A_bf16[M,K] @ Bt_bf16[N,K]^T + bias), f32 accumulate.
// A row-major [M,K]; B given TRANSPOSED (column-major): Bt[n][k], row stride K.
// Block = 256 threads = 8 waves. Wave tile = (TM*16) x (TN*16).
// Double-buffered LDS; staging via GLOBAL_LOAD_ASYNC_TO_LDS_B128 when the
// toolchain exposes it (ASYNCcnt-tracked DMA overlapped with WMMA), else a
// batched synchronous b128 copy. One barrier per K-step.
// ---------------------------------------------------------------------------
template<int BM, int BN, int TM, int TN, bool RELU, bool OUTBF>
__global__ __launch_bounds__(256) void gemm_wmma(
    const unsigned short* __restrict__ A,
    const unsigned short* __restrict__ Bt,
    const float* __restrict__ bias,
    void* __restrict__ Cout,
    int M, int N, int K)
{
    constexpr int BK = 32;
    __shared__ __align__(16) unsigned short As[2][BM][BK + 8];
    __shared__ __align__(16) unsigned short Bs[2][BN][BK + 8];

    const int tid  = threadIdx.x;
    const int lane = tid & 31;
    const int hl   = lane >> 4;   // lane half (0/1)
    const int lr   = lane & 15;   // lane row
    const int wave = tid >> 5;
    constexpr int NWN = BN / (TN * 16);   // waves along N
    static_assert((BM / (TM * 16)) * NWN == 8, "8 waves per block");
    const int wrow = (wave / NWN) * (TM * 16);
    const int wcol = (wave % NWN) * (TN * 16);
    const int rowA0 = blockIdx.y * BM;
    const int colB0 = blockIdx.x * BN;

    constexpr int AITER = (BM * BK) / (256 * 8);
    constexpr int BITER = (BN * BK) / (256 * 8);

    auto stage = [&](int buf, int k0) {
#if defined(HAVE_ASYNC_LDS)
#pragma unroll
        for (int it = 0; it < AITER; ++it) {
            int idx = tid + it * 256;
            int r = idx >> 2, c = (idx & 3) << 3;
            __builtin_amdgcn_global_load_async_to_lds_b128(
                (GAS1 v4i*)(A + (size_t)(rowA0 + r) * K + (k0 + c)),
                (LAS3 v4i*)(&As[buf][r][c]), 0, 0);
        }
#pragma unroll
        for (int it = 0; it < BITER; ++it) {
            int idx = tid + it * 256;
            int r = idx >> 2, c = (idx & 3) << 3;
            __builtin_amdgcn_global_load_async_to_lds_b128(
                (GAS1 v4i*)(Bt + (size_t)(colB0 + r) * K + (k0 + c)),
                (LAS3 v4i*)(&Bs[buf][r][c]), 0, 0);
        }
#else
        uint4 ta[AITER], tb[BITER];
#pragma unroll
        for (int it = 0; it < AITER; ++it) {
            int idx = tid + it * 256;
            int r = idx >> 2, c = (idx & 3) << 3;
            ta[it] = *(const uint4*)(A + (size_t)(rowA0 + r) * K + (k0 + c));
        }
#pragma unroll
        for (int it = 0; it < BITER; ++it) {
            int idx = tid + it * 256;
            int r = idx >> 2, c = (idx & 3) << 3;
            tb[it] = *(const uint4*)(Bt + (size_t)(colB0 + r) * K + (k0 + c));
        }
#pragma unroll
        for (int it = 0; it < AITER; ++it) {
            int idx = tid + it * 256;
            int r = idx >> 2, c = (idx & 3) << 3;
            *(uint4*)(&As[buf][r][c]) = ta[it];
        }
#pragma unroll
        for (int it = 0; it < BITER; ++it) {
            int idx = tid + it * 256;
            int r = idx >> 2, c = (idx & 3) << 3;
            *(uint4*)(&Bs[buf][r][c]) = tb[it];
        }
#endif
    };

    v8f acc[TM][TN];
#pragma unroll
    for (int i = 0; i < TM; ++i)
#pragma unroll
        for (int j = 0; j < TN; ++j)
            acc[i][j] = v8f{0.f,0.f,0.f,0.f,0.f,0.f,0.f,0.f};

    stage(0, 0);
    int cur = 0;
    for (int k0 = 0; k0 < K; k0 += BK) {
        wait_async0();        // my async LDS writes for buffer `cur` have landed
        __syncthreads();      // everyone's writes visible; prior reads of `cur^1` done
        if (k0 + BK < K) stage(cur ^ 1, k0 + BK);   // overlap next tile with compute

        Frag16 af[TM], bfr[TN];
#pragma unroll
        for (int mi = 0; mi < TM; ++mi)
#pragma unroll
            for (int v = 0; v < 8; ++v) {
                int kb = ((v >= 4) ? 16 : 0) + hl * 8 + (v & 3) * 2;
                af[mi].u[v] = *(const unsigned int*)(&As[cur][wrow + mi * 16 + lr][kb]);
            }
#pragma unroll
        for (int ni = 0; ni < TN; ++ni)
#pragma unroll
            for (int v = 0; v < 8; ++v) {
                int kb = ((v >= 4) ? 16 : 0) + hl * 8 + (v & 3) * 2;
                bfr[ni].u[v] = *(const unsigned int*)(&Bs[cur][wcol + ni * 16 + lr][kb]);
            }
#pragma unroll
        for (int mi = 0; mi < TM; ++mi)
#pragma unroll
            for (int ni = 0; ni < TN; ++ni)
                acc[mi][ni] = __builtin_amdgcn_wmma_f32_16x16x32_bf16(
                    false, af[mi].v, false, bfr[ni].v,
                    (short)0, acc[mi][ni], false, false);
        cur ^= 1;
    }

    // -------- epilogue: bias + relu, f32 or bf16 store --------
#pragma unroll
    for (int mi = 0; mi < TM; ++mi) {
#pragma unroll
        for (int ni = 0; ni < TN; ++ni) {
            int col = colB0 + wcol + ni * 16 + lr;
            float bia = bias ? bias[col] : 0.f;
#pragma unroll
            for (int r = 0; r < 8; ++r) {
                int row = rowA0 + wrow + mi * 16 + r + hl * 8;
                float val = acc[mi][ni][r] + bia;
                if (RELU) val = fmaxf(val, 0.f);
                if (OUTBF)
                    ((unsigned short*)Cout)[(size_t)row * N + col] = f2bf(val);
                else
                    ((float*)Cout)[(size_t)row * N + col] = val;
            }
        }
    }
}

// ---------------------------------------------------------------------------
// Tiled transpose + f32->bf16 convert: src[R][C] f32 -> dst[C][R] bf16.
// ---------------------------------------------------------------------------
__global__ __launch_bounds__(256) void k_cvt_t(const float* __restrict__ src,
                                               unsigned short* __restrict__ dst,
                                               int R, int C) {
    __shared__ float tile[32][33];
    int rb = blockIdx.y * 32, cb = blockIdx.x * 32;
    int tx = threadIdx.x & 31, ty = threadIdx.x >> 5;  // 32 x 8
#pragma unroll
    for (int i = ty; i < 32; i += 8)
        tile[i][tx] = src[(size_t)(rb + i) * C + (cb + tx)];
    __syncthreads();
#pragma unroll
    for (int i = ty; i < 32; i += 8)
        dst[(size_t)(cb + i) * R + (rb + tx)] = f2bf(tile[tx][i]);
}

// degree -> rsqrt : A = clip(D + D^T, 0, 1) + I
__global__ void k_dinv(const int* __restrict__ D, float* __restrict__ dinv) {
    int i = blockIdx.x;
    __shared__ float red[256];
    float s = 0.f;
    for (int j = threadIdx.x; j < TT; j += 256) {
        int a = D[(size_t)i * TT + j] + D[(size_t)j * TT + i];
        float v = (a > 0) ? 1.f : 0.f;
        if (j == i) v += 1.f;
        s += v;
    }
    red[threadIdx.x] = s; __syncthreads();
    for (int o = 128; o > 0; o >>= 1) {
        if (threadIdx.x < o) red[threadIdx.x] += red[threadIdx.x + o];
        __syncthreads();
    }
    if (threadIdx.x == 0) dinv[i] = rsqrtf(red[0]);
}

// An[i][j] = dinv[i]*dinv[j]*(clip(Dij+Dji,0,1) + [i==j])  -> bf16
__global__ void k_an(const int* __restrict__ D, const float* __restrict__ dinv,
                     unsigned short* __restrict__ An) {
    int i = blockIdx.x;
    float di = dinv[i];
    for (int j = threadIdx.x; j < TT; j += 256) {
        int a = D[(size_t)i * TT + j] + D[(size_t)j * TT + i];
        float v = ((a > 0) ? 1.f : 0.f) + ((j == i) ? 1.f : 0.f);
        An[(size_t)i * TT + j] = f2bf(v * di * dinv[j]);
    }
}

// tf = x @ Wtp + btp     ([T,2] @ [2,16])
__global__ void k_tf(const float* __restrict__ x, const float* __restrict__ W,
                     const float* __restrict__ b, float* __restrict__ tf) {
    int idx = blockIdx.x * 256 + threadIdx.x;
    if (idx >= TT * NFEAT) return;
    int i = idx / NFEAT, f = idx % NFEAT;
    tf[idx] = x[i * 2] * W[f] + x[i * 2 + 1] * W[NFEAT + f] + b[f];
}

// pf = x^T @ Wpp + bpp   ([2,T] @ [T,16])
__global__ void k_pf(const float* __restrict__ x, const float* __restrict__ W,
                     const float* __restrict__ b, float* __restrict__ pf) {
    int t = threadIdx.x; if (t >= 2 * NFEAT) return;
    int p = t >> 4, f = t & 15;
    float s = b[f];
    for (int i = 0; i < TT; ++i) s += x[i * 2 + p] * W[i * NFEAT + f];
    pf[p * NFEAT + f] = s;
}

// Zt[c][i] = bf16((H @ W + b)[i][c]); rows c >= Creal zero-padded (Cout rows)
__global__ void k_lin_bf16_t(const float* __restrict__ Hin, const float* __restrict__ W,
                             const float* __restrict__ b, unsigned short* __restrict__ Zt,
                             int K, int Cout, int Creal) {
    int idx = blockIdx.x * 256 + threadIdx.x;
    if (idx >= TT * Cout) return;
    int c = idx / TT, i = idx % TT;
    float s = 0.f;
    if (c < Creal) {
        s = b[c];
        for (int k = 0; k < K; ++k) s += Hin[(size_t)i * K + k] * W[k * Creal + c];
    }
    Zt[(size_t)c * TT + i] = f2bf(s);
}

// column mean over T rows
__global__ void k_colmean(const float* __restrict__ Min, float* __restrict__ mean,
                          int C, int ld) {
    int c = blockIdx.x * blockDim.x + threadIdx.x; if (c >= C) return;
    float s = 0.f;
    for (int i = 0; i < TT; ++i) s += Min[(size_t)i * ld + c];
    mean[c] = s * (1.f / TT);
}

// mean of the 2 provider rows
__global__ void k_mean2(const float* __restrict__ rows, float* __restrict__ out, int C) {
    int c = threadIdx.x;
    if (c < C) out[c] = 0.5f * (rows[c] + rows[C + c]);
}

// neighbor terms: nbT = meanP@Wn + bn ; nbP = meanT@Wn + bn
__global__ void k_nb(const float* __restrict__ meanP, const float* __restrict__ meanT,
                     const float* __restrict__ Wn, const float* __restrict__ bn,
                     float* __restrict__ nbT, float* __restrict__ nbP, int K, int C) {
    int c = blockIdx.x * blockDim.x + threadIdx.x; if (c >= C) return;
    float s1 = bn[c], s2 = bn[c];
    for (int k = 0; k < K; ++k) {
        float w = Wn[k * C + c];
        s1 += meanP[k] * w; s2 += meanT[k] * w;
    }
    nbT[c] = s1; nbP[c] = s2;
}

// task rows of GraphConv layer: relu(H@Ws + bs + nbT)
__global__ void k_gc_tasks(const float* __restrict__ Hin, const float* __restrict__ Ws,
                           const float* __restrict__ bs, const float* __restrict__ nbT,
                           float* __restrict__ Gout, int K, int C) {
    int idx = blockIdx.x * 256 + threadIdx.x;
    if (idx >= TT * C) return;
    int i = idx / C, c = idx % C;
    float s = bs[c] + nbT[c];
    for (int k = 0; k < K; ++k) s += Hin[(size_t)i * K + k] * Ws[k * C + c];
    Gout[idx] = fmaxf(s, 0.f);
}

// provider rows (layer 1 only)
__global__ void k_gc_prov(const float* __restrict__ pf, const float* __restrict__ Ws,
                          const float* __restrict__ bs, const float* __restrict__ nbP,
                          float* __restrict__ Gprov, int K, int C) {
    int t = threadIdx.x; if (t >= 2 * C) return;
    int p = t / C, c = t % C;
    float s = bs[c] + nbP[c];
    for (int k = 0; k < K; ++k) s += pf[p * K + k] * Ws[k * C + c];
    Gprov[p * C + c] = fmaxf(s, 0.f);
}

// fused = bf16(concat(task_gnn[:, :32], g2_tasks[:, :32]))  (A of GEMM1, row-major)
__global__ void k_fuse(const float* __restrict__ h3, const float* __restrict__ g2t,
                       unsigned short* __restrict__ fb) {
    int idx = blockIdx.x * 256 + threadIdx.x;
    if (idx >= TT * 64) return;
    int i = idx / 64, c = idx % 64;
    float v = (c < 32) ? h3[(size_t)i * 64 + c] : g2t[(size_t)i * 32 + (c - 32)];
    fb[idx] = f2bf(v);
}

// ---------------------------------------------------------------------------
extern "C" void kernel_launch(void* const* d_in, const int* in_sizes, int n_in,
                              void* d_out, int out_size, void* d_ws, size_t ws_size,
                              hipStream_t stream) {
    (void)in_sizes; (void)n_in; (void)out_size; (void)ws_size;
    const float* x   = (const float*)d_in[0];
    const int*   D   = (const int*)  d_in[1];
    // d_in[2] edge_index: structure is known complete-bipartite -> unused
    const float* Wtp = (const float*)d_in[3];
    const float* btp = (const float*)d_in[4];
    const float* Wpp = (const float*)d_in[5];
    const float* bpp = (const float*)d_in[6];
    const float* Wg1 = (const float*)d_in[7];
    const float* bg1 = (const float*)d_in[8];
    const float* Wg2 = (const float*)d_in[9];
    const float* bg2 = (const float*)d_in[10];
    const float* Wg3 = (const float*)d_in[11];
    const float* bg3 = (const float*)d_in[12];
    const float* Ws1 = (const float*)d_in[13];
    const float* bs1 = (const float*)d_in[14];
    const float* Wn1 = (const float*)d_in[15];
    const float* bn1 = (const float*)d_in[16];
    const float* Ws2 = (const float*)d_in[17];
    const float* bs2 = (const float*)d_in[18];
    const float* Wn2 = (const float*)d_in[19];
    const float* bn2 = (const float*)d_in[20];
    const float* Wf1 = (const float*)d_in[21];
    const float* bf1 = (const float*)d_in[22];
    const float* Wf2 = (const float*)d_in[23];
    const float* bf2 = (const float*)d_in[24];

    char* w = (char*)d_ws;
    auto alloc = [&](size_t bytes) -> void* {
        void* p = (void*)w;
        w += (bytes + 255) & ~(size_t)255;
        return p;
    };
    unsigned short* Anb   = (unsigned short*)alloc((size_t)TT * TT * 2);       //  32 MB (A, row-major)
    unsigned short* Wf2bT = (unsigned short*)alloc((size_t)FDIM * FDIM * 2);   // 128 MB (B^T [N][K])
    unsigned short* Wf1bT = (unsigned short*)alloc((size_t)FDIM * 64 * 2);     //   1 MB (B^T [8192][64])
    unsigned short* hidb  = (unsigned short*)alloc((size_t)TT * FDIM * 2);     //  64 MB (A of GEMM2)
    float* tf   = (float*)alloc((size_t)TT * NFEAT * 4);
    float* pf   = (float*)alloc(256);
    float* dinv = (float*)alloc((size_t)TT * 4);
    unsigned short* Zt = (unsigned short*)alloc((size_t)64 * TT * 2);          // B^T [64][T]
    float* h1  = (float*)alloc((size_t)TT * 64 * 4);
    float* h2  = (float*)alloc((size_t)TT * 64 * 4);
    float* h3  = (float*)alloc((size_t)TT * 64 * 4);
    float* g1t = (float*)alloc((size_t)TT * 64 * 4);
    float* g1p = (float*)alloc(2 * 64 * 4);
    float* g2t = (float*)alloc((size_t)TT * 32 * 4);
    unsigned short* fb = (unsigned short*)alloc((size_t)TT * 64 * 2);
    float* sm = (float*)alloc(4096);
    float *meanT1 = sm,       *meanP1 = sm + 32,  *nbT1 = sm + 64,  *nbP1 = sm + 128;
    float *meanT2 = sm + 192, *meanP2 = sm + 256, *nbT2 = sm + 320, *nbP2 = sm + 384;

    // bf16 transposed weight conversions (Wf2^T stays L2-resident: 128 MB < 192 MB)
    k_cvt_t<<<dim3(FDIM / 32, FDIM / 32), 256, 0, stream>>>(Wf2, Wf2bT, FDIM, FDIM);
    k_cvt_t<<<dim3(FDIM / 32, 64 / 32),   256, 0, stream>>>(Wf1, Wf1bT, 64, FDIM);

    // normalized adjacency (dinv folded in) as bf16
    k_dinv<<<TT, 256, 0, stream>>>(D, dinv);
    k_an<<<TT, 256, 0, stream>>>(D, dinv, Anb);

    // feature projections
    k_tf<<<(TT * NFEAT + 255) / 256, 256, 0, stream>>>(x, Wtp, btp, tf);
    k_pf<<<1, 32, 0, stream>>>(x, Wpp, bpp, pf);

    // --- 3-layer task GCN: h = relu(An @ (h@W + b)) via WMMA ---
    dim3 gprop(64 / 64, TT / 128);
    k_lin_bf16_t<<<(TT * 64 + 255) / 256, 256, 0, stream>>>(tf, Wg1, bg1, Zt, NFEAT, 64, 64);
    gemm_wmma<128, 64, 2, 2, true, false><<<gprop, 256, 0, stream>>>(Anb, Zt, nullptr, h1, TT, 64, TT);
    k_lin_bf16_t<<<(TT * 64 + 255) / 256, 256, 0, stream>>>(h1, Wg2, bg2, Zt, 64, 64, 64);
    gemm_wmma<128, 64, 2, 2, true, false><<<gprop, 256, 0, stream>>>(Anb, Zt, nullptr, h2, TT, 64, TT);
    k_lin_bf16_t<<<(TT * 64 + 255) / 256, 256, 0, stream>>>(h2, Wg3, bg3, Zt, 64, 64, GOUT);
    gemm_wmma<128, 64, 2, 2, true, false><<<gprop, 256, 0, stream>>>(Anb, Zt, nullptr, h3, TT, 64, TT);

    // --- bipartite GraphConv (complete graph => neighbor means are broadcasts) ---
    k_colmean<<<1, 64, 0, stream>>>(tf, meanT1, NFEAT, NFEAT);
    k_mean2<<<1, 64, 0, stream>>>(pf, meanP1, NFEAT);
    k_nb<<<1, 64, 0, stream>>>(meanP1, meanT1, Wn1, bn1, nbT1, nbP1, NFEAT, 64);
    k_gc_tasks<<<(TT * 64 + 255) / 256, 256, 0, stream>>>(tf, Ws1, bs1, nbT1, g1t, NFEAT, 64);
    k_gc_prov<<<1, 128, 0, stream>>>(pf, Ws1, bs1, nbP1, g1p, NFEAT, 64);
    k_colmean<<<1, 64, 0, stream>>>(g1t, meanT2, 64, 64);
    k_mean2<<<1, 64, 0, stream>>>(g1p, meanP2, 64);
    k_nb<<<1, 64, 0, stream>>>(meanP2, meanT2, Wn2, bn2, nbT2, nbP2, 64, GOUT);
    k_gc_tasks<<<(TT * GOUT + 255) / 256, 256, 0, stream>>>(g1t, Ws2, bs2, nbT2, g2t, 64, GOUT);

    // --- fusion + big WMMA GEMMs (compute-bound; Wf2bT L2-resident) ---
    k_fuse<<<(TT * 64 + 255) / 256, 256, 0, stream>>>(h3, g2t, fb);
    dim3 gbig(FDIM / 128, TT / 128);
    gemm_wmma<128, 128, 2, 4, true, true ><<<gbig, 256, 0, stream>>>(fb,   Wf1bT, bf1, hidb,  TT, FDIM, 64);
    gemm_wmma<128, 128, 2, 4, true, false><<<gbig, 256, 0, stream>>>(hidb, Wf2bT, bf2, d_out, TT, FDIM, FDIM);
}
